// MultiTaskLoss_80522046865975
// MI455X (gfx1250) — matrix operand order repaired and checked by reading, not compiled, not run on hardware
//
#include <hip/hip_runtime.h>
#include <hip/hip_bf16.h>
#include <math.h>

typedef __attribute__((ext_vector_type(16))) _Float16 v16h;
typedef __attribute__((ext_vector_type(8)))  float    v8f;

#define NUM_ANT 16
#define PI_F    3.14159265358979323846f
// RHO = 0.8 : comm = -2.0*min - 0.2*sum_user ; sens = -0.05*sum_target

// ---------------------------------------------------------------------------
// Kernel A: angle estimation (phase unwrap + OLS slope) for 8 user + 4 target
// angles, then write TWO WMMA *A* operands (f16, 16x32 each) pre-swizzled into
// the A register layout (ISA 7.12.2, 16-bit A 16x32):
//   lane L: row M = L&15; element e: K = e + (L>=16 ? 16 : 0) (contiguous, same as B; not ISA table order)
// A0 rows: M 0..7  = user-re(angle m), M 8..15 = user-im(angle m-8)
// A1 rows: M 0..3  = target-re,        M 8..11 = target-im, rest zero
// Coefficient(combo, k): k<16 multiplies Wr, k>=16 multiplies Wi:
//   re: k<16 -> cos(ph[k]),  k>=16 -> sin(ph[k-16])    (re = Wr@c + Wi@s)
//   im: k<16 -> -sin(ph[k]), k>=16 -> cos(ph[k-16])    (im = Wi@c - Wr@s)
// Stored as packed half-pairs: dword (t*8 + e/2)*32 + lane = elems (e, e+1).
// ---------------------------------------------------------------------------
__global__ __launch_bounds__(32)
void prep_kernel(const float* __restrict__ Hc_r, const float* __restrict__ Hc_i,
                 const float* __restrict__ Hs_r, const float* __restrict__ Hs_i,
                 _Float16* __restrict__ Aws) {
  __shared__ float sh_sin[12];   // sin(deg2rad(angle)) per angle
  const int tid = threadIdx.x;
  if (tid < 12) {
    const float* hr; const float* hi;
    if (tid < 8) { hr = Hc_r + tid * NUM_ANT;       hi = Hc_i + tid * NUM_ANT; }
    else         { hr = Hs_r + (tid - 8) * NUM_ANT; hi = Hs_i + (tid - 8) * NUM_ANT; }
    float prev = atan2f(hi[0], hr[0]);
    float unw = 0.f, acc = 0.f;
    for (int n = 1; n < NUM_ANT; ++n) {
      float p  = atan2f(hi[n], hr[n]);
      float pd = p - prev; prev = p;
      float m  = fmodf(pd + PI_F, 2.f * PI_F);
      if (m < 0.f) m += 2.f * PI_F;     // python-style mod
      m -= PI_F;
      unw += m;
      acc += unw * ((float)n - 7.5f);
    }
    float slope = acc * (1.0f / 340.0f);              // sum(n_c^2) = 340
    float st = slope / (2.f * PI_F * 0.5f);           // * lambda / (2 pi D)
    st = fminf(1.f, fmaxf(-1.f, st));
    sh_sin[tid] = sinf(-asinf(st));                   // deg conversions cancel
  }
  __syncthreads();

  const int lane = tid;
  const int m    = lane & 15;                // combo row
  const int kup  = (lane >= 16) ? 16 : 0;    // A layout upper-lane K offset
  for (int t = 0; t < 2; ++t) {
#pragma unroll
    for (int e = 0; e < 16; ++e) {
      const int k = e + kup;   // 0..31
      float val = 0.f;
      int a = -1; bool isIm = false;
      if (t == 0)                      { a = m & 7;       isIm = (m >= 8); }
      else if ((m & 7) < 4)            { a = 8 + (m & 3); isIm = (m >= 8); }
      if (a >= 0) {
        const int ant = (k < 16) ? k : (k - 16);
        const float ph = 2.f * PI_F * 0.5f * sh_sin[a] * (float)ant;
        float s, c; sincosf(ph, &s, &c);
        val = isIm ? ((k < 16) ? -s : c)
                   : ((k < 16) ?  c : s);
      }
      Aws[((t * 8 + (e >> 1)) * 32 + lane) * 2 + (e & 1)] = (_Float16)val;
    }
  }
}

// ---------------------------------------------------------------------------
// Kernel B: bandwidth-bound streaming GEMM + cheap epilogue.
// B operand = 16-row W tile: lane n holds row n (dup n+16), K half per
// lane-half, 64 contiguous bytes per lane. D: VGPR v = combo v (lanes 0..15)
// and combo v+8 (lanes 16..31), N = lane&15 = W row.
// gain_v(row) = sqrt(D[v]^2 + shfl_xor16(D[v]^2)); min over v is per-lane.
// Sum terms are linear -> accumulated per lane, reduced once per wave.
// ---------------------------------------------------------------------------
__global__ __launch_bounds__(256)
void gemm_kernel(const float* __restrict__ W, const _Float16* __restrict__ Aws,
                 float* __restrict__ partials, int nTiles) {
  const int lane = threadIdx.x & 31;

  union AU { v16h h; unsigned u[8]; };
  AU a0, a1;
  const unsigned* aw = (const unsigned*)Aws;
#pragma unroll
  for (int j = 0; j < 8; ++j) {
    a0.u[j] = aw[j * 32 + lane];
    a1.u[j] = aw[256 + j * 32 + lane];
  }

  const int waveId = (int)((blockIdx.x * blockDim.x + threadIdx.x) >> 5);
  const int nWaves = (int)((gridDim.x * blockDim.x) >> 5);
  const int koff   = (lane >= 16) ? 16 : 0;   // B layout K-half select
  float accm = 0.f;   // sum over rows of min user gain (per lane-dup)
  float accg = 0.f;   // sum of all user gains
  float acct = 0.f;   // sum of all target gains

  for (int tile = waveId; tile < nTiles; tile += nWaves) {
    const float* rowp = W + ((size_t)tile * 16 + (size_t)(lane & 15)) * 32 + koff;
    if (tile + nWaves < nTiles)
      __builtin_prefetch(rowp + (size_t)nWaves * 16 * 32, 0, 0);

    union { float4 v[4]; float f[16]; } cw;
    cw.v[0] = *(const float4*)(rowp);
    cw.v[1] = *(const float4*)(rowp + 4);
    cw.v[2] = *(const float4*)(rowp + 8);
    cw.v[3] = *(const float4*)(rowp + 12);

    v16h b;
#pragma unroll
    for (int e = 0; e < 16; ++e) b[e] = (_Float16)cw.f[e];

    v8f cz = {};
    v8f d0 = __builtin_amdgcn_wmma_f32_16x16x32_f16(false, a0.h, false, b, (short)0, cz, false, false);
    v8f d1 = __builtin_amdgcn_wmma_f32_16x16x32_f16(false, a1.h, false, b, (short)0, cz, false, false);

    float gmin;
#pragma unroll
    for (int v = 0; v < 8; ++v) {
      float sq = d0[v] * d0[v];
      float s2 = sq + __shfl_xor(sq, 16, 32);     // re^2 + im^2, both halves
      float g  = __builtin_amdgcn_sqrtf(s2);
      gmin = (v == 0) ? g : fminf(gmin, g);
      accg += g;
    }
    accm += gmin;
#pragma unroll
    for (int v = 0; v < 4; ++v) {
      float sq = d1[v] * d1[v];
      float s2 = sq + __shfl_xor(sq, 16, 32);
      acct += __builtin_amdgcn_sqrtf(s2);
    }
  }

  // Each W row is duplicated on lanes n and n+16 -> halve at the end.
  float local = -2.0f * accm - 0.2f * accg - 0.05f * acct;
#pragma unroll
  for (int m = 1; m <= 16; m <<= 1) local += __shfl_xor(local, m, 32);

  __shared__ float sh[8];
  if (lane == 0) sh[threadIdx.x >> 5] = 0.5f * local;
  __syncthreads();
  if (threadIdx.x == 0) {
    float s = 0.f;
#pragma unroll
    for (int i = 0; i < 8; ++i) s += sh[i];
    partials[blockIdx.x] = s;       // every block writes -> deterministic
  }
}

// ---------------------------------------------------------------------------
// Kernel C: deterministic final reduction -> mean -> d_out[0]
// ---------------------------------------------------------------------------
__global__ __launch_bounds__(256)
void reduce_kernel(const float* __restrict__ partials, int n, float invRows,
                   float* __restrict__ out) {
  __shared__ float sh[256];
  float s = 0.f;
  for (int i = threadIdx.x; i < n; i += 256) s += partials[i];
  sh[threadIdx.x] = s;
  __syncthreads();
  for (int k = 128; k > 0; k >>= 1) {
    if (threadIdx.x < k) sh[threadIdx.x] += sh[threadIdx.x + k];
    __syncthreads();
  }
  if (threadIdx.x == 0) out[0] = sh[0] * invRows;
}

extern "C" void kernel_launch(void* const* d_in, const int* in_sizes, int n_in,
                              void* d_out, int out_size, void* d_ws, size_t ws_size,
                              hipStream_t stream) {
  const float* W    = (const float*)d_in[0];   // [nrows, 32]
  const float* Hc_r = (const float*)d_in[1];   // [4, 8, 16]
  const float* Hc_i = (const float*)d_in[2];
  const float* Hs_r = (const float*)d_in[3];   // [4, 4, 16]
  const float* Hs_i = (const float*)d_in[4];
  float* out = (float*)d_out;

  const int nrows  = in_sizes[0] / 32;     // 2,000,000
  const int nTiles = nrows / 16;           // 125,000

  _Float16* Aws      = (_Float16*)d_ws;                     // 2048 bytes
  float*    partials = (float*)((char*)d_ws + 2048);        // 4096 bytes

  const int blocks = 1024;

  prep_kernel<<<1, 32, 0, stream>>>(Hc_r, Hc_i, Hs_r, Hs_i, Aws);
  gemm_kernel<<<blocks, 256, 0, stream>>>(W, Aws, partials, nTiles);
  reduce_kernel<<<1, 256, 0, stream>>>(partials, blocks, 1.0f / (float)nrows, out);
}
